// RVB_21766894256832
// MI455X (gfx1250) — matrix operand order repaired and checked by reading, not compiled, not run on hardware
//
#include <hip/hip_runtime.h>

// ---------------------------------------------------------------------------
// RVB mean-field amplitude kernel for MI455X (gfx1250, wave32, WMMA).
//   out[s] = (W/2) * x_s^T chi x_s  +  sum_i log(phi[i, x_si>0 ? 0 : 1])
// Formulated as GEMM Y = X * chi (f16 inputs exact: x in {+-1}, chi in {0,1})
// with fused X (.) Y row-sum + log-phi lookup epilogue.
// ---------------------------------------------------------------------------

typedef _Float16 v16h __attribute__((ext_vector_type(16)));
typedef _Float16 v8h  __attribute__((ext_vector_type(8)));
typedef float    v8f  __attribute__((ext_vector_type(8)));

#define N_SITES 432
#define KPAD    448          // K padded to 14 * 32
#define KSTEPS  14
#define NTILES  27           // 432 / 16 output-column tiles
#define GTILES  3            // n-tiles staged per LDS slab
#define NGROUPS 9            // 27 / 3
#define WAVES   8
#define BLK_M   (WAVES * 16) // 128 samples per block
#define NS_TOT  32768

// --------------------------- prep kernel -----------------------------------
// chi (f32, symmetric) -> chi_b[n][k] f16, K zero-padded to 448.
// phi (N,2) -> lg0[i]=log(phi[i][0]), lg1[i]=log(phi[i][1]).
__global__ void rvb_prep_kernel(const float* __restrict__ chi,
                                const float* __restrict__ phi,
                                _Float16* __restrict__ chi_b,
                                float* __restrict__ lg0,
                                float* __restrict__ lg1) {
  int idx = blockIdx.x * blockDim.x + threadIdx.x;
  if (idx < N_SITES * KPAD) {
    int k = idx % KPAD;
    int n = idx / KPAD;
    float v = (k < N_SITES) ? chi[n * N_SITES + k] : 0.0f;
    chi_b[idx] = (_Float16)v;
  }
  int j = idx - N_SITES * KPAD;
  if (j >= 0 && j < N_SITES) {
    lg0[j] = logf(phi[2 * j + 0]);
    lg1[j] = logf(phi[2 * j + 1]);
  }
}

// --------------------------- main kernel -----------------------------------
__global__ void __launch_bounds__(256)
rvb_wmma_kernel(const float* __restrict__ x,
                const _Float16* __restrict__ chi_b,
                const float* __restrict__ Wp,
                const float* __restrict__ lg0,
                const float* __restrict__ lg1,
                float* __restrict__ out) {
  __shared__ __align__(16) _Float16 slab[GTILES * 16 * KPAD]; // 43008 B
  __shared__ float lgs0[N_SITES];                             // 1728 B
  __shared__ float lgs1[N_SITES];                             // 1728 B

  const int tid   = threadIdx.x;
  const int wave  = tid >> 5;
  const int lane  = tid & 31;
  const int lhal  = lane >> 4;   // 0 or 1 (which 16-lane half)
  const int l15   = lane & 15;
  const int khalf = lhal * 8;    // A-layout K offset per lane-half

  // Stage log-phi tables into LDS (covered by first group barrier).
  for (int i = tid; i < N_SITES; i += 256) {
    lgs0[i] = lg0[i];
    lgs1[i] = lg1[i];
  }

  const int m_base = blockIdx.x * BLK_M + wave * 16;

  // ---- Load the wave's 16x448 f16 A strip into registers (WMMA A layout).
  // A 16x32 f16: lane (l&15)=row M; lane-half selects K {0..7,16..23} vs
  // {8..15,24..31}; element e: k = khalf + (e<8 ? e : 16 + (e-8)).
  const float* xrow = x + (size_t)(m_base + l15) * N_SITES;
  v16h A[KSTEPS];
#pragma unroll
  for (int ks = 0; ks < KSTEPS; ++ks) {
    const int k0 = ks * 32 + khalf;
    float4 f0 = *(const float4*)(xrow + k0);
    float4 f1 = *(const float4*)(xrow + k0 + 4);
    v16h a;
    a[0] = (_Float16)f0.x;  a[1] = (_Float16)f0.y;
    a[2] = (_Float16)f0.z;  a[3] = (_Float16)f0.w;
    a[4] = (_Float16)f1.x;  a[5] = (_Float16)f1.y;
    a[6] = (_Float16)f1.z;  a[7] = (_Float16)f1.w;
    if (ks < KSTEPS - 1) {          // ks==13 upper half is K=432..447 pad
      float4 f2 = *(const float4*)(xrow + k0 + 16);
      float4 f3 = *(const float4*)(xrow + k0 + 20);
      a[8]  = (_Float16)f2.x;  a[9]  = (_Float16)f2.y;
      a[10] = (_Float16)f2.z;  a[11] = (_Float16)f2.w;
      a[12] = (_Float16)f3.x;  a[13] = (_Float16)f3.y;
      a[14] = (_Float16)f3.z;  a[15] = (_Float16)f3.w;
    } else {
      a[8]  = (_Float16)0.0f;  a[9]  = (_Float16)0.0f;
      a[10] = (_Float16)0.0f;  a[11] = (_Float16)0.0f;
      a[12] = (_Float16)0.0f;  a[13] = (_Float16)0.0f;
      a[14] = (_Float16)0.0f;  a[15] = (_Float16)0.0f;
    }
    A[ks] = a;
  }

  float q[8]  = {0.f, 0.f, 0.f, 0.f, 0.f, 0.f, 0.f, 0.f};
  float mf[8] = {0.f, 0.f, 0.f, 0.f, 0.f, 0.f, 0.f, 0.f};

  for (int g = 0; g < NGROUPS; ++g) {
    __syncthreads();  // previous group's slab readers done
    // Stage chi slab for n-tiles [g*3, g*3+3): 48 cols x 448 K of f16.
    {
      const uint4* __restrict__ src =
          (const uint4*)(chi_b + (size_t)g * (GTILES * 16 * KPAD));
      uint4* dst = (uint4*)slab;
      for (int i = tid; i < (GTILES * 16 * KPAD) / 8; i += 256) dst[i] = src[i];
    }
    __syncthreads();

#pragma unroll
    for (int t = 0; t < GTILES; ++t) {
      // B 32x16 f16 layout (matches A): lane (l&15)=col N; element e:
      // k = 8*lhal + (e<8 ? e : 16 + (e-8))  -> two 8-f16 runs per lane.
      // chi symmetric: B[k][n] = chi[k][n] = chi_b[n][k].
      const _Float16* bp = slab + (size_t)(t * 16 + l15) * KPAD;
      v8f acc = {};
#pragma unroll
      for (int ks = 0; ks < KSTEPS; ++ks) {
        const int kb = ks * 32 + 8 * lhal;
        v8h b0 = *(const v8h*)(bp + kb);
        v8h b1 = *(const v8h*)(bp + kb + 16);
        v16h b = __builtin_shufflevector(b0, b1,
                                         0, 1, 2, 3, 4, 5, 6, 7,
                                         8, 9, 10, 11, 12, 13, 14, 15);
        acc = __builtin_amdgcn_wmma_f32_16x16x32_f16(
            /*neg_a=*/false, A[ks], /*neg_b=*/false, b,
            /*c_mod=*/(short)0, acc, /*reuse_a=*/false, /*reuse_b=*/false);
      }
      // Fused epilogue: C layout lane l, reg v -> row = v + 8*lhal, col = l15.
      const int col = (g * GTILES + t) * 16 + l15;
      const float l0v = lgs0[col];
      const float l1v = lgs1[col];
#pragma unroll
      for (int v = 0; v < 8; ++v) {
        const float xv = x[(size_t)(m_base + v + 8 * lhal) * N_SITES + col];
        q[v]  += acc[v] * xv;
        mf[v] += (xv > 0.0f) ? l0v : l1v;
      }
    }
  }

  // Reduce over the 16 lanes of each half (masks < 16 stay within the half).
#pragma unroll
  for (int v = 0; v < 8; ++v) {
#pragma unroll
    for (int m = 1; m <= 8; m <<= 1) {
      q[v]  += __shfl_xor(q[v], m, 32);
      mf[v] += __shfl_xor(mf[v], m, 32);
    }
  }

  const float w = Wp[0];
  if (l15 == 0) {
#pragma unroll
    for (int v = 0; v < 8; ++v) {
      out[m_base + 8 * lhal + v] = 0.5f * w * q[v] + mf[v];
    }
  }
}

// --------------------------- launcher --------------------------------------
extern "C" void kernel_launch(void* const* d_in, const int* in_sizes, int n_in,
                              void* d_out, int out_size, void* d_ws, size_t ws_size,
                              hipStream_t stream) {
  const float* x   = (const float*)d_in[0];   // (32768, 432) f32 in {-1,+1}
  const float* chi = (const float*)d_in[1];   // (432, 432) f32
  const float* W   = (const float*)d_in[2];   // scalar f32
  const float* phi = (const float*)d_in[3];   // (432, 2) f32

  // Workspace layout: chi_b f16 [432*448], lg0 f32 [432], lg1 f32 [432].
  _Float16* chi_b = (_Float16*)d_ws;
  float* lg0 = (float*)((char*)d_ws + (size_t)N_SITES * KPAD * sizeof(_Float16));
  float* lg1 = lg0 + N_SITES;

  const int prep_elems  = N_SITES * KPAD + N_SITES;
  const int prep_blocks = (prep_elems + 255) / 256;
  rvb_prep_kernel<<<prep_blocks, 256, 0, stream>>>(chi, phi, chi_b, lg0, lg1);

  const int main_blocks = NS_TOT / BLK_M; // 256
  rvb_wmma_kernel<<<main_blocks, 256, 0, stream>>>(x, chi_b, W, lg0, lg1,
                                                   (float*)d_out);
}
